// PointCloudTokenizer_50852412785430
// MI455X (gfx1250) — compile-verified, matrix operand order checked by reading, not run on hardware
//
#include <hip/hip_runtime.h>

// ---------------- problem constants ----------------
#define BB 8
#define NN 16384
#define FDIM 6
#define MM 128
#define KK 16
#define DD 768
#define NWAVES 8   // 256 threads = 8 wave32
#define CPB 4      // centroids (16-row tiles) per point-MLP block

typedef __attribute__((ext_vector_type(16))) __bf16 v16bf;
typedef __attribute__((ext_vector_type(8)))  float  v8f;

// ---------------- workspace layout (bytes) ----------------
// bf16 transposed+K-padded weights: WT[n*Kpad + k]
static constexpr size_t OFF_W0T  = 0;                       // 256 x 32
static constexpr size_t OFF_W1T  = OFF_W0T  + 256*32*2;     // 512 x 256
static constexpr size_t OFF_W2T  = OFF_W1T  + 512*256*2;    // 768 x 512
static constexpr size_t OFF_W3T  = OFF_W2T  + 768*512*2;    // 768 x 768
static constexpr size_t OFF_Wn0T = OFF_W3T  + 768*768*2;    // 768 x 768
static constexpr size_t OFF_Wn1T = OFF_Wn0T + 768*768*2;    // 768 x 768
static constexpr size_t OFF_FIDX = OFF_Wn1T + 768*768*2;    // int  [B*M]
static constexpr size_t OFF_CENT = OFF_FIDX + BB*MM*4;      // f32  [B*M*4]
static constexpr size_t OFF_KNN  = OFF_CENT + BB*MM*4*4;    // int  [B*M*K]
static constexpr size_t OFF_POOL = OFF_KNN  + BB*MM*KK*4;   // f32  [B*M*D]
static constexpr size_t OFF_TOK  = OFF_POOL + (size_t)BB*MM*DD*4; // f32 [B*M*D]
static constexpr size_t OFF_ORD  = OFF_TOK  + (size_t)BB*MM*DD*4; // int [B*M]

// ---------------- WMMA fragment loaders ----------------
// A fragment: 16x32 bf16 tile from LDS activations (row stride = KPAD).
// ISA 7.12.2: lane L holds row (L&15); K chunk {0..7,16..23} for lanes 0-15,
// {8..15,24..31} for lanes 16-31 -> two contiguous 16B reads per lane.
template <int KPAD>
__device__ __forceinline__ v16bf load_a_frag(const __bf16* act, int kk, int lane) {
    const __bf16* p = act + (lane & 15) * KPAD + kk * 32 + ((lane & 16) ? 8 : 0);
    v16bf a;
#pragma unroll
    for (int e = 0; e < 8; ++e) a[e] = p[e];
#pragma unroll
    for (int e = 0; e < 8; ++e) a[e + 8] = p[16 + e];
    return a;
}

// B fragment from pre-transposed weights WT[Nout x KPAD] (row stride KPAD):
// lane holds column n; same K indexing as A -> two contiguous 16B global reads.
template <int KPAD>
__device__ __forceinline__ v16bf load_b_frag(const __bf16* WT, int n, int kk, int lane) {
    const __bf16* p = WT + (size_t)n * KPAD + kk * 32 + ((lane & 16) ? 8 : 0);
    v16bf b;
#pragma unroll
    for (int e = 0; e < 8; ++e) b[e] = p[e];
#pragma unroll
    for (int e = 0; e < 8; ++e) b[e + 8] = p[16 + e];
    return b;
}

__device__ __forceinline__ v8f wmma_bf16(v16bf a, v16bf b, v8f c) {
    return __builtin_amdgcn_wmma_f32_16x16x32_bf16(false, a, false, b, (short)0,
                                                   c, false, false);
}

// Epilogues. MODE 0: relu -> bf16 LDS rows; MODE 1: 16-row maxpool -> global
// vector of NOUT; MODE 2: +bias -> f32 global rows.
template <int MODE, int NOUT>
__device__ __forceinline__ void epilogue(v8f acc, int n, const float* __restrict__ bias,
                                         int hi, int lane, __bf16* outLds,
                                         float* outGlob) {
    const float bn = bias[n];
    if (MODE == 0) {
#pragma unroll
        for (int r = 0; r < 8; ++r) {
            float v = fmaxf(acc[r] + bn, 0.f);
            outLds[(r + hi) * NOUT + n] = (__bf16)v;   // native cvt path
        }
    } else if (MODE == 1) {
        float mx = -3.4e38f;
#pragma unroll
        for (int r = 0; r < 8; ++r) mx = fmaxf(mx, acc[r] + bn);
        mx = fmaxf(mx, __shfl_xor(mx, 16, 32));        // combine row halves
        if (!(lane & 16)) outGlob[n] = mx;
    } else {
#pragma unroll
        for (int r = 0; r < 8; ++r)
            outGlob[(size_t)(r + hi) * NOUT + n] = acc[r] + bn;
    }
}

// GEMM over RT 16-row tiles x NOUT, spread over 8 waves. Each wave owns a
// contiguous run of N-tile pairs; per K-step two B fragments are loaded once
// and reused across all RT A fragments (8 WMMAs per B load at RT=4), which
// amortizes the dominant L2 weight traffic.
// MODE 1: outGlob = pooled base; row-tile rt pools into outGlob + rt*DD.
template <int MODE, int KPAD, int NOUT, int RT>
__device__ __forceinline__ void gemm16(const __bf16* __restrict__ in,
                                       const __bf16* __restrict__ WT,
                                       const float* __restrict__ bias,
                                       __bf16* outLds, float* outGlob) {
    constexpr int nK  = KPAD / 32;
    constexpr int nT  = NOUT / 16;
    constexpr int TPW = nT / NWAVES;          // 2, 4 or 6 (always even)
    const int tid  = threadIdx.x;
    const int lane = tid & 31;
    const int wave = tid >> 5;
    const int col  = lane & 15;
    const int hi   = (lane & 16) ? 8 : 0;
#pragma unroll
    for (int t = 0; t < TPW; t += 2) {
        const int n0 = (wave * TPW + t) * 16 + col;
        const int n1 = n0 + 16;
        v8f acc0[RT], acc1[RT];
#pragma unroll
        for (int rt = 0; rt < RT; ++rt) { acc0[rt] = {}; acc1[rt] = {}; }
#pragma unroll 2
        for (int kk = 0; kk < nK; ++kk) {
            v16bf b0 = load_b_frag<KPAD>(WT, n0, kk, lane);
            v16bf b1 = load_b_frag<KPAD>(WT, n1, kk, lane);
#pragma unroll
            for (int rt = 0; rt < RT; ++rt) {
                v16bf a = load_a_frag<KPAD>(in + rt * 16 * KPAD, kk, lane);
                acc0[rt] = wmma_bf16(a, b0, acc0[rt]);
                acc1[rt] = wmma_bf16(a, b1, acc1[rt]);
            }
        }
#pragma unroll
        for (int rt = 0; rt < RT; ++rt) {
            __bf16* ol = (MODE == 0) ? outLds + rt * 16 * NOUT : nullptr;
            float*  og = (MODE == 1) ? outGlob + (size_t)rt * DD
                        : (MODE == 2) ? outGlob + (size_t)rt * 16 * NOUT : nullptr;
            epilogue<MODE, NOUT>(acc0[rt], n0, bias, hi, lane, ol, og);
            epilogue<MODE, NOUT>(acc1[rt], n1, bias, hi, lane, ol, og);
        }
    }
}

// ---------------- kernel 1: weight convert (f32 -> bf16, transposed, K-padded)
__global__ void wconv_kernel(const float* __restrict__ W, __bf16* __restrict__ WT,
                             int Kin, int Kpad, int Nout) {
    int idx = blockIdx.x * 256 + threadIdx.x;
    if (idx >= Kpad * Nout) return;
    int n = idx / Kpad, k = idx % Kpad;
    WT[idx] = (k < Kin) ? (__bf16)W[(size_t)k * Nout + n] : (__bf16)0.f;
}

// ---------------- kernel 2: farthest point sampling (one block per batch)
__global__ __launch_bounds__(1024) void fps_kernel(const float* __restrict__ coords,
                                                   int* __restrict__ fidx,
                                                   float* __restrict__ cent) {
    const int b = blockIdx.x, tid = threadIdx.x;
    const float* base = coords + (size_t)b * NN * 5;
    float px[16], py[16], pz[16], pt[16], mind[16];
#pragma unroll
    for (int t = 0; t < 16; ++t) {
        int i = tid + t * 1024;
        px[t] = base[i * 5 + 1]; py[t] = base[i * 5 + 2];
        pz[t] = base[i * 5 + 3]; pt[t] = base[i * 5 + 4];
        mind[t] = 1e30f;
    }
    __shared__ float s_val[32];
    __shared__ int   s_idx[32];
    __shared__ int   s_last;
    if (tid == 0) {
        s_last = 0;
        fidx[b * MM] = 0;
        cent[(b * MM) * 4 + 0] = base[1]; cent[(b * MM) * 4 + 1] = base[2];
        cent[(b * MM) * 4 + 2] = base[3]; cent[(b * MM) * 4 + 3] = base[4];
    }
    __syncthreads();
    for (int it = 1; it < MM; ++it) {
        const int last = s_last;
        const float cx = base[last * 5 + 1], cy = base[last * 5 + 2];
        const float cz = base[last * 5 + 3], ct = base[last * 5 + 4];
        float bestv = -1.f; int besti = 0x7fffffff;
#pragma unroll
        for (int t = 0; t < 16; ++t) {
            float dx = px[t] - cx, dy = py[t] - cy, dz = pz[t] - cz, dw = pt[t] - ct;
            float d = dx * dx + dy * dy + dz * dz + dw * dw;
            mind[t] = fminf(mind[t], d);
            if (mind[t] > bestv) { bestv = mind[t]; besti = tid + t * 1024; }
        }
#pragma unroll
        for (int off = 16; off > 0; off >>= 1) {
            float ov = __shfl_down(bestv, off, 32);
            int   oi = __shfl_down(besti, off, 32);
            if (ov > bestv || (ov == bestv && oi < besti)) { bestv = ov; besti = oi; }
        }
        if ((tid & 31) == 0) { s_val[tid >> 5] = bestv; s_idx[tid >> 5] = besti; }
        __syncthreads();
        if (tid < 32) {
            bestv = s_val[tid]; besti = s_idx[tid];
#pragma unroll
            for (int off = 16; off > 0; off >>= 1) {
                float ov = __shfl_down(bestv, off, 32);
                int   oi = __shfl_down(besti, off, 32);
                if (ov > bestv || (ov == bestv && oi < besti)) { bestv = ov; besti = oi; }
            }
            if (tid == 0) {
                s_last = besti;
                fidx[b * MM + it] = besti;
                cent[(b * MM + it) * 4 + 0] = base[besti * 5 + 1];
                cent[(b * MM + it) * 4 + 1] = base[besti * 5 + 2];
                cent[(b * MM + it) * 4 + 2] = base[besti * 5 + 3];
                cent[(b * MM + it) * 4 + 3] = base[besti * 5 + 4];
            }
        }
        __syncthreads();
    }
}

// ---------------- kernel 3: kNN (one block per (b,m)); top-16 via local sorted
// lists + log-tree merge in LDS
__global__ __launch_bounds__(256) void knn_kernel(const float* __restrict__ coords,
                                                  const float* __restrict__ cent,
                                                  int* __restrict__ knn) {
    const int bm = blockIdx.x, b = bm / MM, tid = threadIdx.x;
    const float* base = coords + (size_t)b * NN * 5;
    const float cx = cent[bm * 4 + 0], cy = cent[bm * 4 + 1];
    const float cz = cent[bm * 4 + 2], ct = cent[bm * 4 + 3];
    float ld[KK]; int li[KK];
#pragma unroll
    for (int k = 0; k < KK; ++k) { ld[k] = 1e30f; li[k] = 0x7fffffff; }
    for (int i = tid; i < NN; i += 256) {
        float dx = base[i * 5 + 1] - cx, dy = base[i * 5 + 2] - cy;
        float dz = base[i * 5 + 3] - cz, dw = base[i * 5 + 4] - ct;
        float d = dx * dx + dy * dy + dz * dz + dw * dw;
        if (d < ld[KK - 1] || (d == ld[KK - 1] && i < li[KK - 1])) {
            int j = KK - 1;
            while (j > 0 && (ld[j - 1] > d || (ld[j - 1] == d && li[j - 1] > i))) {
                ld[j] = ld[j - 1]; li[j] = li[j - 1]; --j;
            }
            ld[j] = d; li[j] = i;
        }
    }
    __shared__ float sd[256 * KK];
    __shared__ int   si[256 * KK];
#pragma unroll
    for (int k = 0; k < KK; ++k) { sd[tid * KK + k] = ld[k]; si[tid * KK + k] = li[k]; }
    __syncthreads();
    for (int s = 128; s > 0; s >>= 1) {
        if (tid < s) {
            float od[KK], nd[KK]; int oi[KK], ni[KK];
#pragma unroll
            for (int k = 0; k < KK; ++k) { od[k] = sd[(tid + s) * KK + k]; oi[k] = si[(tid + s) * KK + k]; }
            int p = 0, q = 0;
#pragma unroll
            for (int k = 0; k < KK; ++k) {
                bool takeMine = (ld[p] < od[q]) || (ld[p] == od[q] && li[p] < oi[q]);
                if (takeMine) { nd[k] = ld[p]; ni[k] = li[p]; ++p; }
                else          { nd[k] = od[q]; ni[k] = oi[q]; ++q; }
            }
#pragma unroll
            for (int k = 0; k < KK; ++k) {
                ld[k] = nd[k]; li[k] = ni[k];
                sd[tid * KK + k] = nd[k]; si[tid * KK + k] = ni[k];
            }
        }
        __syncthreads();
    }
    if (tid < KK) knn[bm * KK + tid] = si[tid];
}

// ---------------- kernel 4: fused per-point MLP on gathered neighbors + maxpool
// one block per CPB=4 centroids: 64 neighbor rows through 6->256->512->768->768.
// LDS: bufA 64x768 bf16 (96KB) + bufB 64x512 bf16 (64KB) = 160KB.
__global__ __launch_bounds__(256) void point_mlp_kernel(
    const float* __restrict__ feats, const int* __restrict__ knn,
    const __bf16* __restrict__ W0T, const float* __restrict__ b0,
    const __bf16* __restrict__ W1T, const float* __restrict__ b1,
    const __bf16* __restrict__ W2T, const float* __restrict__ b2,
    const __bf16* __restrict__ W3T, const float* __restrict__ b3,
    float* __restrict__ pooled) {
    __shared__ __bf16 bufA[CPB * 16 * 768];   // 96 KB
    __shared__ __bf16 bufB[CPB * 16 * 512];   // 64 KB
    const int bm0 = blockIdx.x * CPB;
    const int b   = bm0 / MM;
    const int tid = threadIdx.x;
    // stage 64 neighbor feature rows into bufB, K-padded 6 -> 32
    for (int e = tid; e < CPB * 16 * 32; e += 256) {
        int row = e >> 5, k = e & 31;
        float v = 0.f;
        if (k < FDIM) {
            int p = knn[(bm0 + (row >> 4)) * KK + (row & 15)];
            v = feats[((size_t)b * NN + p) * FDIM + k];
        }
        bufB[row * 32 + k] = (__bf16)v;
    }
    __syncthreads();
    gemm16<0, 32, 256, CPB>(bufB, W0T, b0, bufA, nullptr);   // bufB(64x32) -> bufA(64x256)
    __syncthreads();
    gemm16<0, 256, 512, CPB>(bufA, W1T, b1, bufB, nullptr);  // bufA(64x256)-> bufB(64x512)
    __syncthreads();
    gemm16<0, 512, 768, CPB>(bufB, W2T, b2, bufA, nullptr);  // bufB(64x512)-> bufA(64x768)
    __syncthreads();
    gemm16<1, 768, 768, CPB>(bufA, W3T, b3, nullptr, pooled + (size_t)bm0 * DD);
}

// ---------------- kernel 5: token MLP (relu(pooled@Wn0+bn0)@Wn1+bn1), WMMA bf16
// one block per 16 rows of the flattened [B*M, D] pooled matrix
__global__ __launch_bounds__(256) void token_mlp_kernel(
    const float* __restrict__ pooled,
    const __bf16* __restrict__ Wn0T, const float* __restrict__ bn0,
    const __bf16* __restrict__ Wn1T, const float* __restrict__ bn1,
    float* __restrict__ tokens) {
    __shared__ __bf16 bufA[16 * 768];
    __shared__ __bf16 bufB[16 * 768];
    const int rowBase = blockIdx.x * 16, tid = threadIdx.x;
    for (int e = tid; e < 16 * 768; e += 256) {
        int r = e / 768, c = e % 768;
        bufA[e] = (__bf16)pooled[(size_t)(rowBase + r) * DD + c];
    }
    __syncthreads();
    gemm16<0, 768, 768, 1>(bufA, Wn0T, bn0, bufB, nullptr);
    __syncthreads();
    gemm16<2, 768, 768, 1>(bufB, Wn1T, bn1, nullptr, tokens + (size_t)rowBase * DD);
}

// ---------------- kernel 6: per-batch bitonic argsort by centroid time + outputs
__global__ __launch_bounds__(128) void sort_kernel(const float* __restrict__ centW,
                                                   int* __restrict__ order,
                                                   float* __restrict__ centOut,
                                                   float* __restrict__ maskOut) {
    const int b = blockIdx.x, tid = threadIdx.x;
    __shared__ float key[MM];
    __shared__ int   idx[MM];
    key[tid] = centW[(b * MM + tid) * 4 + 3];
    idx[tid] = tid;
    __syncthreads();
    for (int k = 2; k <= MM; k <<= 1)
        for (int j = k >> 1; j > 0; j >>= 1) {
            int ixj = tid ^ j;
            if (ixj > tid) {
                bool asc = ((tid & k) == 0);
                float a = key[tid], v = key[ixj];
                int   ia = idx[tid], ib = idx[ixj];
                bool gt = (a > v) || (a == v && ia > ib);  // stable tie-break
                if (gt == asc) { key[tid] = v; key[ixj] = a; idx[tid] = ib; idx[ixj] = ia; }
            }
            __syncthreads();
        }
    int src = idx[tid];
    order[b * MM + tid] = src;
#pragma unroll
    for (int c = 0; c < 4; ++c)
        centOut[(b * MM + tid) * 4 + c] = centW[(b * MM + src) * 4 + c];
    maskOut[b * MM + tid] = 1.0f;
}

// ---------------- kernel 7: gather tokens by sort order
__global__ __launch_bounds__(256) void gather_tokens_kernel(
    const float* __restrict__ tokW, const int* __restrict__ order,
    float* __restrict__ tokOut) {
    const int bm = blockIdx.x, b = bm / MM;
    const int src = order[bm];
    const float* s = tokW + ((size_t)b * MM + src) * DD;
    float* d = tokOut + (size_t)bm * DD;
    for (int i = threadIdx.x; i < DD; i += 256) d[i] = s[i];
}

// ---------------- launch ----------------
extern "C" void kernel_launch(void* const* d_in, const int* in_sizes, int n_in,
                              void* d_out, int out_size, void* d_ws, size_t ws_size,
                              hipStream_t stream) {
    const float* coords = (const float*)d_in[0];
    const float* feats  = (const float*)d_in[1];
    const float* W0 = (const float*)d_in[2];  const float* b0 = (const float*)d_in[3];
    const float* W1 = (const float*)d_in[4];  const float* b1 = (const float*)d_in[5];
    const float* W2 = (const float*)d_in[6];  const float* b2 = (const float*)d_in[7];
    const float* W3 = (const float*)d_in[8];  const float* b3 = (const float*)d_in[9];
    const float* Wn0 = (const float*)d_in[10]; const float* bn0 = (const float*)d_in[11];
    const float* Wn1 = (const float*)d_in[12]; const float* bn1 = (const float*)d_in[13];

    char* ws = (char*)d_ws;
    __bf16* W0T  = (__bf16*)(ws + OFF_W0T);
    __bf16* W1T  = (__bf16*)(ws + OFF_W1T);
    __bf16* W2T  = (__bf16*)(ws + OFF_W2T);
    __bf16* W3T  = (__bf16*)(ws + OFF_W3T);
    __bf16* Wn0T = (__bf16*)(ws + OFF_Wn0T);
    __bf16* Wn1T = (__bf16*)(ws + OFF_Wn1T);
    int*   fidx   = (int*)(ws + OFF_FIDX);
    float* centW  = (float*)(ws + OFF_CENT);
    int*   knn    = (int*)(ws + OFF_KNN);
    float* pooled = (float*)(ws + OFF_POOL);
    float* tokW   = (float*)(ws + OFF_TOK);
    int*   order  = (int*)(ws + OFF_ORD);

    float* tokOut  = (float*)d_out;
    float* centOut = tokOut + (size_t)BB * MM * DD;
    float* maskOut = centOut + (size_t)BB * MM * 4;

    // 1) weight conversion (transpose + K-pad to 32, f32 -> bf16)
    wconv_kernel<<<(256 * 32 + 255) / 256, 256, 0, stream>>>(W0, W0T, FDIM, 32, 256);
    wconv_kernel<<<(512 * 256 + 255) / 256, 256, 0, stream>>>(W1, W1T, 256, 256, 512);
    wconv_kernel<<<(768 * 512 + 255) / 256, 256, 0, stream>>>(W2, W2T, 512, 512, 768);
    wconv_kernel<<<(768 * 768 + 255) / 256, 256, 0, stream>>>(W3, W3T, 768, 768, 768);
    wconv_kernel<<<(768 * 768 + 255) / 256, 256, 0, stream>>>(Wn0, Wn0T, 768, 768, 768);
    wconv_kernel<<<(768 * 768 + 255) / 256, 256, 0, stream>>>(Wn1, Wn1T, 768, 768, 768);
    // 2) FPS centroids
    fps_kernel<<<BB, 1024, 0, stream>>>(coords, fidx, centW);
    // 3) kNN
    knn_kernel<<<BB * MM, 256, 0, stream>>>(coords, centW, knn);
    // 4) fused gather + point MLP + maxpool (WMMA, 4 centroids per block)
    point_mlp_kernel<<<BB * MM / CPB, 256, 0, stream>>>(feats, knn, W0T, b0, W1T, b1,
                                                        W2T, b2, W3T, b3, pooled);
    // 5) token MLP (WMMA)
    token_mlp_kernel<<<BB * MM / 16, 256, 0, stream>>>(pooled, Wn0T, bn0, Wn1T, bn1, tokW);
    // 6) sort by centroid time, emit cent + masks
    sort_kernel<<<BB, MM, 0, stream>>>(centW, order, centOut, maskOut);
    // 7) gather tokens into sorted order
    gather_tokens_kernel<<<BB * MM, 256, 0, stream>>>(tokW, order, tokOut);
}
